// Attention_88897233092854
// MI455X (gfx1250) — matrix operand order repaired and checked by reading, not compiled, not run on hardware
//
#include <hip/hip_runtime.h>
#include <hip/hip_bf16.h>
#include <math.h>
#include <stdint.h>

typedef __attribute__((ext_vector_type(16))) _Float16 v16h;
typedef __attribute__((ext_vector_type(8)))  float    v8f;

union Frag { v16h h; unsigned u[8]; };

#define WMMA_F32_F16(a, b, c) \
    __builtin_amdgcn_wmma_f32_16x16x32_f16(false, (a), false, (b), (short)0, (c), false, false)

// CDNA5 async global->LDS copy (ASYNCcnt-tracked), 16B per lane.
__device__ __forceinline__ void async_copy_b128(const void* gptr, void* lptr) {
    unsigned lds = (unsigned)(uintptr_t)lptr;   // low 32 bits of flat LDS addr = LDS offset
    asm volatile("global_load_async_to_lds_b128 %0, %1, off"
                 :: "v"(lds), "v"(gptr) : "memory");
}
__device__ __forceinline__ void wait_async0() {
    asm volatile("s_wait_asynccnt 0x0" ::: "memory");
}

// ---------------------------------------------------------------------------
// f32 -> f16 convert (grid-stride)
// ---------------------------------------------------------------------------
__global__ void cvt_f16_kernel(const float* __restrict__ src,
                               _Float16* __restrict__ dst, int n) {
    int i = blockIdx.x * blockDim.x + threadIdx.x;
    int stride = gridDim.x * blockDim.x;
    for (; i < n; i += stride) dst[i] = (_Float16)src[i];
}

// ---------------------------------------------------------------------------
// Tiled transpose + convert: dst[N x K] (f16) = src[K x N] (f32) ^T
// 32x32 tiles, 256 threads (32x8), LDS staging with padded stride.
// ---------------------------------------------------------------------------
__launch_bounds__(256)
__global__ void transpose_cvt_kernel(const float* __restrict__ src,
                                     _Float16* __restrict__ dst,
                                     int K, int N) {
    __shared__ float t[32][33];
    const int n0 = blockIdx.x * 32;
    const int k0 = blockIdx.y * 32;
    const int tx = threadIdx.x & 31;
    const int ty = threadIdx.x >> 5;    // 0..7
#pragma unroll
    for (int i = 0; i < 32; i += 8)
        t[ty + i][tx] = src[(size_t)(k0 + ty + i) * N + n0 + tx];
    __syncthreads();
#pragma unroll
    for (int i = 0; i < 32; i += 8)
        dst[(size_t)(n0 + ty + i) * K + k0 + tx] = (_Float16)t[tx][ty + i];
}

// ---------------------------------------------------------------------------
// Tiled WMMA GEMM: C[M,N] = A[M,K] @ Wt[N,K]^T + bias, optional exact GELU.
// 256 threads = 8 waves; 128x128 C tile; wave does 32x64 (2x4 WMMA tiles).
// Both A and B(=Wt rows) tiles staged by async global->LDS B128, double
// buffered: one s_wait_asynccnt + one barrier per 32-wide K step.
// ---------------------------------------------------------------------------
template <int ACT_GELU, int OUT_F16>
__launch_bounds__(256)
__global__ void gemm_kernel(const _Float16* __restrict__ A,
                            const _Float16* __restrict__ Bt,   // [N x K]
                            const float* __restrict__ bias,
                            void* __restrict__ Cout,
                            int Mdim, int Ndim, int Kdim) {
    __shared__ __align__(16) _Float16 As[2][128 * 32];
    __shared__ __align__(16) _Float16 Bs[2][128 * 32];   // n-major: [n*32 + k]

    const int tid  = threadIdx.x;
    const int lane = tid & 31;
    const int wave = tid >> 5;
    const int wm   = wave >> 1;     // 0..3
    const int wn   = wave & 1;      // 0..1
    const int m0   = blockIdx.y * 128;
    const int n0   = blockIdx.x * 128;
    const int lr   = lane & 15;
    const int lh   = lane >> 4;

    const int r  = tid >> 1;          // staging row 0..127 (for both A and Bt)
    const int ks = (tid & 1) * 16;    // staging k-offset

    v8f acc[2][4];
#pragma unroll
    for (int i = 0; i < 2; ++i)
#pragma unroll
        for (int j = 0; j < 4; ++j)
#pragma unroll
            for (int q = 0; q < 8; ++q) acc[i][j][q] = 0.0f;

    auto stage = [&](int buf, int k0) {
        const _Float16* ga = A  + (size_t)(m0 + r) * Kdim + k0 + ks;
        const _Float16* gb = Bt + (size_t)(n0 + r) * Kdim + k0 + ks;
        _Float16* la = &As[buf][r * 32 + ks];
        _Float16* lb = &Bs[buf][r * 32 + ks];
        async_copy_b128(ga,     la);
        async_copy_b128(ga + 8, la + 8);
        async_copy_b128(gb,     lb);
        async_copy_b128(gb + 8, lb + 8);
    };

    const int nsteps = Kdim >> 5;
    stage(0, 0);
    for (int s = 0; s < nsteps; ++s) {
        wait_async0();
        __syncthreads();
        if (s + 1 < nsteps) stage((s + 1) & 1, (s + 1) << 5);
        const int buf = s & 1;

        Frag afr[2], bfr[4];
#pragma unroll
        for (int ms = 0; ms < 2; ++ms) {
            const unsigned* p = (const unsigned*)&As[buf][0] + (wm * 32 + ms * 16 + lr) * 16;
#pragma unroll
            for (int i = 0; i < 4; ++i) {
                afr[ms].u[i]     = p[lh * 4 + i];
                afr[ms].u[4 + i] = p[8 + lh * 4 + i];
            }
        }
#pragma unroll
        for (int ns2 = 0; ns2 < 4; ++ns2) {
            const unsigned* p = (const unsigned*)&Bs[buf][0] + (wn * 64 + ns2 * 16 + lr) * 16;
#pragma unroll
            for (int i = 0; i < 8; ++i) bfr[ns2].u[i] = p[(i >> 2) * 8 + lh * 4 + (i & 3)];
        }
#pragma unroll
        for (int ms = 0; ms < 2; ++ms)
#pragma unroll
            for (int ns2 = 0; ns2 < 4; ++ns2)
                acc[ms][ns2] = WMMA_F32_F16(afr[ms].h, bfr[ns2].h, acc[ms][ns2]);
    }

    // --- epilogue: bias (+gelu), store f16 or f32 ---
#pragma unroll
    for (int ms = 0; ms < 2; ++ms) {
        int rowb = m0 + wm * 32 + ms * 16 + lh * 8;
#pragma unroll
        for (int ns2 = 0; ns2 < 4; ++ns2) {
            int col = n0 + wn * 64 + ns2 * 16 + lr;
            float bb = bias[col];
#pragma unroll
            for (int q = 0; q < 8; ++q) {
                float v = acc[ms][ns2][q] + bb;
                if (ACT_GELU) v = 0.5f * v * (1.0f + erff(v * 0.7071067811865476f));
                size_t idx = (size_t)(rowb + q) * Ndim + col;
                if (OUT_F16) ((_Float16*)Cout)[idx] = (_Float16)v;
                else         ((float*)Cout)[idx]    = v;
            }
        }
    }
}

// ---------------------------------------------------------------------------
// Flash attention, causal, H=16 heads of 64, L=2048, f16 QKV -> f32 out.
// grid = (L/128, B*H); 8 waves; wave w owns query rows [qtile*128 + w*16, +16).
// K tile staged with async global->LDS; V tile transposed via VGPR scatter.
// ---------------------------------------------------------------------------
__launch_bounds__(256)
__global__ void flash_attn_kernel(const _Float16* __restrict__ qh,
                                  const _Float16* __restrict__ kh,
                                  const _Float16* __restrict__ vh,
                                  float* __restrict__ attn_out) {
    constexpr int L = 2048, D = 1024;
    __shared__ __align__(16) _Float16 Ks[128 * 64];    // key-major, d contiguous
    __shared__ __align__(16) _Float16 Vt[64 * 128];    // d-major, key contiguous
    __shared__ __align__(16) _Float16 Ps[8 * 16 * 128];

    const int tid  = threadIdx.x;
    const int lane = tid & 31;
    const int wave = tid >> 5;
    const int lr   = lane & 15;
    const int lh   = lane >> 4;
    const int qtile = blockIdx.x;
    const int bh    = blockIdx.y;
    const int b     = bh >> 4;
    const int hh    = bh & 15;

    // Q fragments for this wave's 16 rows (two k-chunks of 32 over hd=64)
    Frag qa[2];
#pragma unroll
    for (int kc = 0; kc < 2; ++kc) {
        int grow = b * L + qtile * 128 + wave * 16 + lr;
        const unsigned* p = (const unsigned*)qh + (size_t)grow * 512 + (hh * 64 + kc * 32) / 2;
#pragma unroll
        for (int i = 0; i < 4; ++i) {
            qa[kc].u[i]     = p[lh * 4 + i];
            qa[kc].u[4 + i] = p[8 + lh * 4 + i];
        }
    }

    float mrow[8], lrow[8];
    v8f o[4];
#pragma unroll
    for (int q = 0; q < 8; ++q) { mrow[q] = -1e30f; lrow[q] = 0.0f; }
#pragma unroll
    for (int t = 0; t < 4; ++t)
#pragma unroll
        for (int q = 0; q < 8; ++q) o[t][q] = 0.0f;

    _Float16* pw = Ps + wave * 2048;

    for (int kt = 0; kt <= qtile; ++kt) {
        // --- stage K tile (async) and transposed V tile (scatter) ---
        {
            int ky = tid >> 1;
            int ds = (tid & 1) * 32;
            size_t grow = (size_t)(b * L + kt * 128 + ky) * D + hh * 64 + ds;
            const _Float16* gk = kh + grow;
            _Float16* sk = Ks + ky * 64 + ds;
            async_copy_b128(gk,      sk);
            async_copy_b128(gk + 8,  sk + 8);
            async_copy_b128(gk + 16, sk + 16);
            async_copy_b128(gk + 24, sk + 24);
            union { uint4 q[4]; _Float16 e[32]; } tv;
            const uint4* gv = (const uint4*)(vh + grow);
            tv.q[0] = gv[0]; tv.q[1] = gv[1]; tv.q[2] = gv[2]; tv.q[3] = gv[3];
#pragma unroll
            for (int i = 0; i < 32; ++i) Vt[(ds + i) * 128 + ky] = tv.e[i];
        }
        wait_async0();
        __syncthreads();

        // --- S = Q K^T (16 x 128) ---
        v8f s[8];
#pragma unroll
        for (int t = 0; t < 8; ++t)
#pragma unroll
            for (int q = 0; q < 8; ++q) s[t][q] = 0.0f;
#pragma unroll
        for (int kc = 0; kc < 2; ++kc) {
#pragma unroll
            for (int t = 0; t < 8; ++t) {
                Frag bf;
                const unsigned* p = (const unsigned*)Ks + (t * 16 + lr) * 32 + kc * 16;
#pragma unroll
                for (int i = 0; i < 8; ++i) bf.u[i] = p[(i >> 2) * 8 + lh * 4 + (i & 3)];
                s[t] = WMMA_F32_F16(qa[kc].h, bf.h, s[t]);
            }
        }

        // --- scale, causal mask, online softmax ---
        const bool diag = (kt == qtile);
        const int qrow0 = qtile * 128 + wave * 16 + lh * 8;
        const int col0  = kt * 128 + lr;
        float mx[8];
#pragma unroll
        for (int q = 0; q < 8; ++q) mx[q] = -1e30f;
#pragma unroll
        for (int t = 0; t < 8; ++t)
#pragma unroll
            for (int q = 0; q < 8; ++q) {
                float v = s[t][q] * 0.125f;
                if (diag && (col0 + t * 16) > (qrow0 + q)) v = -1000.0f;
                s[t][q] = v;
                mx[q] = fmaxf(mx[q], v);
            }
#pragma unroll
        for (int q = 0; q < 8; ++q)
#pragma unroll
            for (int off = 1; off < 16; off <<= 1)
                mx[q] = fmaxf(mx[q], __shfl_xor(mx[q], off, 32));

        float corr[8], psum[8];
#pragma unroll
        for (int q = 0; q < 8; ++q) {
            float mn = fmaxf(mrow[q], mx[q]);
            corr[q] = expf(mrow[q] - mn);
            mrow[q] = mn;
            psum[q] = 0.0f;
        }
#pragma unroll
        for (int t = 0; t < 8; ++t)
#pragma unroll
            for (int q = 0; q < 8; ++q) {
                float p = expf(s[t][q] - mrow[q]);
                s[t][q] = p;
                psum[q] += p;
            }
#pragma unroll
        for (int q = 0; q < 8; ++q)
#pragma unroll
            for (int off = 1; off < 16; off <<= 1)
                psum[q] += __shfl_xor(psum[q], off, 32);
#pragma unroll
        for (int q = 0; q < 8; ++q) lrow[q] = lrow[q] * corr[q] + psum[q];
#pragma unroll
        for (int t = 0; t < 4; ++t)
#pragma unroll
            for (int q = 0; q < 8; ++q) o[t][q] *= corr[q];

        // --- write P (C layout) into this wave's LDS region, row-major f16 ---
#pragma unroll
        for (int t = 0; t < 8; ++t)
#pragma unroll
            for (int q = 0; q < 8; ++q)
                pw[(lh * 8 + q) * 128 + t * 16 + lr] = (_Float16)s[t][q];

        // --- O += P @ V (16x128 @ 128x64), 4 k-chunks ---
#pragma unroll
        for (int kc2 = 0; kc2 < 4; ++kc2) {
            Frag af;
            const unsigned* ap = (const unsigned*)pw + lr * 64 + kc2 * 16;
#pragma unroll
            for (int i = 0; i < 4; ++i) {
                af.u[i]     = ap[lh * 4 + i];
                af.u[4 + i] = ap[8 + lh * 4 + i];
            }
#pragma unroll
            for (int t = 0; t < 4; ++t) {
                Frag bf;
                const unsigned* bp = (const unsigned*)Vt + (t * 16 + lr) * 64 + kc2 * 16;
#pragma unroll
                for (int i = 0; i < 8; ++i) bf.u[i] = bp[(i >> 2) * 8 + lh * 4 + (i & 3)];
                o[t] = WMMA_F32_F16(af.h, bf.h, o[t]);
            }
        }
        __syncthreads();   // tiles may be overwritten next iteration
    }

    // --- normalize and store f32 ---
#pragma unroll
    for (int t = 0; t < 4; ++t) {
        int col = hh * 64 + t * 16 + lr;
#pragma unroll
        for (int q = 0; q < 8; ++q) {
            int row = qtile * 128 + wave * 16 + lh * 8 + q;
            attn_out[(size_t)(b * L + row) * D + col] = o[t][q] / lrow[q];
        }
    }
}

// ---------------------------------------------------------------------------
// out = LayerNorm(a + b) * gamma + beta; writes f32 and optional f16 copy.
// ---------------------------------------------------------------------------
__launch_bounds__(256)
__global__ void add_ln_kernel(const float* __restrict__ A,
                              const float* __restrict__ Bv,
                              const float* __restrict__ gamma,
                              const float* __restrict__ beta,
                              float* __restrict__ out32,
                              _Float16* __restrict__ out16) {
    __shared__ float red[256];
    const int row = blockIdx.x;
    const int tid = threadIdx.x;
    const float* pa = A  + (size_t)row * 1024;
    const float* pb = Bv + (size_t)row * 1024;
    float x[4];
    float s = 0.0f;
#pragma unroll
    for (int i = 0; i < 4; ++i) { x[i] = pa[tid + 256 * i] + pb[tid + 256 * i]; s += x[i]; }
    red[tid] = s;
    __syncthreads();
    for (int st = 128; st > 0; st >>= 1) {
        if (tid < st) red[tid] += red[tid + st];
        __syncthreads();
    }
    float mu = red[0] * (1.0f / 1024.0f);
    __syncthreads();
    float s2 = 0.0f;
#pragma unroll
    for (int i = 0; i < 4; ++i) { float d = x[i] - mu; s2 += d * d; }
    red[tid] = s2;
    __syncthreads();
    for (int st = 128; st > 0; st >>= 1) {
        if (tid < st) red[tid] += red[tid + st];
        __syncthreads();
    }
    float rstd = rsqrtf(red[0] * (1.0f / 1024.0f) + 1e-5f);
#pragma unroll
    for (int i = 0; i < 4; ++i) {
        int c = tid + 256 * i;
        float y = (x[i] - mu) * rstd * gamma[c] + beta[c];
        out32[(size_t)row * 1024 + c] = y;
        if (out16) out16[(size_t)row * 1024 + c] = (_Float16)y;
    }
}

// ---------------------------------------------------------------------------
extern "C" void kernel_launch(void* const* d_in, const int* in_sizes, int n_in,
                              void* d_out, int out_size, void* d_ws, size_t ws_size,
                              hipStream_t stream) {
    const float* x   = (const float*)d_in[0];
    const float* wq  = (const float*)d_in[1];
    const float* bq  = (const float*)d_in[2];
    const float* wk  = (const float*)d_in[3];
    const float* bk  = (const float*)d_in[4];
    const float* wv  = (const float*)d_in[5];
    const float* bv  = (const float*)d_in[6];
    const float* w1  = (const float*)d_in[7];
    const float* b1  = (const float*)d_in[8];
    const float* w2  = (const float*)d_in[9];
    const float* b2  = (const float*)d_in[10];
    const float* lng = (const float*)d_in[11];
    const float* lnb = (const float*)d_in[12];

    constexpr int Bn = 2, L = 2048, D = 1024, H = 16;
    constexpr int M = Bn * L;          // 4096 rows
    constexpr int F = 4 * D;           // 4096 ffn

    char* base = (char*)d_ws;
    size_t off = 0;
    auto carve = [&](size_t bytes) -> void* {
        void* p = base + off;
        off += (bytes + 255) & ~(size_t)255;
        return p;
    };
    _Float16* xh   = (_Float16*)carve((size_t)M * D * 2);
    _Float16* wqT  = (_Float16*)carve((size_t)D * D * 2);   // [N=D x K=D]
    _Float16* wkT  = (_Float16*)carve((size_t)D * D * 2);
    _Float16* wvT  = (_Float16*)carve((size_t)D * D * 2);
    _Float16* w1T  = (_Float16*)carve((size_t)D * F * 2);   // [N=F x K=D]
    _Float16* w2T  = (_Float16*)carve((size_t)F * D * 2);   // [N=D x K=F]
    _Float16* qh   = (_Float16*)carve((size_t)M * D * 2);
    _Float16* kh   = (_Float16*)carve((size_t)M * D * 2);
    _Float16* vh   = (_Float16*)carve((size_t)M * D * 2);
    float*    attn = (float*)   carve((size_t)M * D * 4);
    float*    h32  = (float*)   carve((size_t)M * D * 4);
    _Float16* h16  = (_Float16*)carve((size_t)M * D * 2);
    _Float16* hid  = (_Float16*)carve((size_t)M * F * 2);
    float*    mlp  = (float*)   carve((size_t)M * D * 4);

    dim3 blk(256);

    // --- precision conversion (+ weight transposition to n-major) ---
    cvt_f16_kernel<<<2048, 256, 0, stream>>>(x, xh, M * D);
    transpose_cvt_kernel<<<dim3(D / 32, D / 32), blk, 0, stream>>>(wq, wqT, D, D);
    transpose_cvt_kernel<<<dim3(D / 32, D / 32), blk, 0, stream>>>(wk, wkT, D, D);
    transpose_cvt_kernel<<<dim3(D / 32, D / 32), blk, 0, stream>>>(wv, wvT, D, D);
    transpose_cvt_kernel<<<dim3(F / 32, D / 32), blk, 0, stream>>>(w1, w1T, D, F);
    transpose_cvt_kernel<<<dim3(D / 32, F / 32), blk, 0, stream>>>(w2, w2T, F, D);

    // --- QKV projections (f16 out for attention WMMA) ---
    dim3 gqkv(D / 128, M / 128);
    gemm_kernel<0, 1><<<gqkv, blk, 0, stream>>>(xh, wqT, bq, qh, M, D, D);
    gemm_kernel<0, 1><<<gqkv, blk, 0, stream>>>(xh, wkT, bk, kh, M, D, D);
    gemm_kernel<0, 1><<<gqkv, blk, 0, stream>>>(xh, wvT, bv, vh, M, D, D);

    // --- causal flash attention ---
    flash_attn_kernel<<<dim3(L / 128, Bn * H), blk, 0, stream>>>(qh, kh, vh, attn);

    // --- h = LN(x + attn) ---
    add_ln_kernel<<<M, blk, 0, stream>>>(x, attn, lng, lnb, h32, h16);

    // --- MLP ---
    gemm_kernel<1, 1><<<dim3(F / 128, M / 128), blk, 0, stream>>>(h16, w1T, b1, hid, M, F, D);
    gemm_kernel<0, 0><<<dim3(D / 128, M / 128), blk, 0, stream>>>(hid, w2T, b2, mlp, M, D, F);

    // --- out = LN(h + mlp) ---
    add_ln_kernel<<<M, blk, 0, stream>>>(h32, mlp, lng, lnb, (float*)d_out, nullptr);
}
